// RWKVLayer_12249246729059
// MI455X (gfx1250) — matrix operand order repaired and checked by reading, not compiled, not run on hardware
//
#include <hip/hip_runtime.h>
#include <hip/hip_bf16.h>

// ---------------- problem constants (fixed by the reference) ----------------
static constexpr int Bk  = 8;
static constexpr int Tk  = 2048;
static constexpr int Ck  = 1024;
static constexpr int HID = 4096;
static constexpr int BT  = Bk * Tk;            // 16384 rows

typedef __attribute__((ext_vector_type(16))) __bf16 bfrag16;
typedef __attribute__((ext_vector_type(8)))  __bf16 bhalf8;
typedef __attribute__((ext_vector_type(8)))  float  v8f;
typedef __attribute__((ext_vector_type(4)))  int    v4i;

// fp32 -> bf16 round-to-nearest-even (bit pattern)
__device__ __forceinline__ unsigned short f2bf(float f) {
    unsigned u = __float_as_uint(f);
    u += 0x7FFFu + ((u >> 16) & 1u);
    return (unsigned short)(u >> 16);
}

// ---------------------------------------------------------------------------
// async global -> LDS copy (16B per lane), CDNA5 GLOBAL_LOAD_ASYNC_TO_LDS_B128
// builtin signature (from probe): (global v4i*, shared v4i*, imm offset, imm cpol)
// ---------------------------------------------------------------------------
#if defined(__has_builtin)
#if __has_builtin(__builtin_amdgcn_global_load_async_to_lds_b128)
#define ASYNC_LDS_BUILTIN 1
#endif
#if __has_builtin(__builtin_amdgcn_s_wait_asynccnt)
#define ASYNC_WAIT_BUILTIN 1
#endif
#endif

__device__ __forceinline__ void async_cp16(unsigned short* lds_dst,
                                           const unsigned short* g_src) {
#if defined(ASYNC_LDS_BUILTIN)
    __builtin_amdgcn_global_load_async_to_lds_b128(
        (__attribute__((address_space(1))) v4i*)(size_t)g_src,
        (__attribute__((address_space(3))) v4i*)(unsigned)(size_t)lds_dst,
        0, 0);
#else
    unsigned lds_off = (unsigned)(size_t)lds_dst;   // flat LDS addr low 32 bits
    asm volatile("global_load_async_to_lds_b128 %0, %1, off"
                 :: "v"(lds_off), "v"(g_src) : "memory");
#endif
}

template<int N>
__device__ __forceinline__ void wait_asynccnt() {
#if defined(ASYNC_WAIT_BUILTIN)
    __builtin_amdgcn_s_wait_asynccnt(N);
#else
    asm volatile("s_wait_asynccnt %0" :: "i"(N) : "memory");
#endif
}

// ---------------------------------------------------------------------------
// fp32 -> bf16 conversion (weights)
// ---------------------------------------------------------------------------
__global__ __launch_bounds__(256) void cvt_f32_bf16(const float* __restrict__ src,
                                                    unsigned short* __restrict__ dst,
                                                    int n) {
    int i = blockIdx.x * 256 + threadIdx.x;
    if (i < n) dst[i] = f2bf(src[i]);
}

// ---------------------------------------------------------------------------
// block reduction helper (256 threads)
// ---------------------------------------------------------------------------
__device__ __forceinline__ float block_sum(float v, float* sm) {
    int t = threadIdx.x;
    sm[t] = v;
    __syncthreads();
    #pragma unroll
    for (int s = 128; s > 0; s >>= 1) {
        if (t < s) sm[t] += sm[t + s];
        __syncthreads();
    }
    float r = sm[0];
    __syncthreads();
    return r;
}

// ---------------------------------------------------------------------------
// Fused LayerNorm + time-shift + token-mix; one block per (b,t) row.
// N_OUT = 3 (att: xk,xv,xr) or 2 (ffn: fk,fr).
// xx[t] = LN(x[t-1]) for t>0, else 0.
// ---------------------------------------------------------------------------
template<int N_OUT>
__global__ __launch_bounds__(256) void ln_mix_kernel(
    const float* __restrict__ x,
    const float* __restrict__ g, const float* __restrict__ bta,
    const float* __restrict__ m0, const float* __restrict__ m1,
    const float* __restrict__ m2,
    unsigned short* __restrict__ o0, unsigned short* __restrict__ o1,
    unsigned short* __restrict__ o2)
{
    __shared__ float sm[256];
    const int row = blockIdx.x;            // 0..BT-1
    const int t   = row & (Tk - 1);
    const int tid = threadIdx.x;
    const float* cur = x + (size_t)row * Ck;
    const float* prv = cur - Ck;           // only read when t > 0

    float c4[4], p4[4];
    #pragma unroll
    for (int i = 0; i < 4; ++i) {
        int ci = i * 256 + tid;
        c4[i] = cur[ci];
        p4[i] = (t > 0) ? prv[ci] : 0.0f;
    }
    float s = 0.f, ss = 0.f;
    #pragma unroll
    for (int i = 0; i < 4; ++i) { s += c4[i]; ss += c4[i] * c4[i]; }
    s  = block_sum(s,  sm);
    ss = block_sum(ss, sm);
    float mean = s * (1.0f / Ck);
    float inv  = rsqrtf(ss * (1.0f / Ck) - mean * mean + 1e-5f);

    float sp = 0.f, ssp = 0.f;
    #pragma unroll
    for (int i = 0; i < 4; ++i) { sp += p4[i]; ssp += p4[i] * p4[i]; }
    sp  = block_sum(sp,  sm);
    ssp = block_sum(ssp, sm);
    float meanp = sp * (1.0f / Ck);
    float invp  = rsqrtf(ssp * (1.0f / Ck) - meanp * meanp + 1e-5f);

    #pragma unroll
    for (int i = 0; i < 4; ++i) {
        int ci = i * 256 + tid;
        float gg = g[ci], bb = bta[ci];
        float xa = (c4[i] - mean) * inv * gg + bb;
        float xx = (t > 0) ? ((p4[i] - meanp) * invp * gg + bb) : 0.0f;
        size_t idx = (size_t)row * Ck + ci;
        float k0 = m0[ci];
        o0[idx] = f2bf(xa * k0 + xx * (1.0f - k0));
        float k1 = m1[ci];
        o1[idx] = f2bf(xa * k1 + xx * (1.0f - k1));
        if (N_OUT == 3) {
            float k2 = m2[ci];
            o2[idx] = f2bf(xa * k2 + xx * (1.0f - k2));
        }
    }
}

// ---------------------------------------------------------------------------
// WKV scan: one thread per (b, channel). Sequential over T.
// Fuses sigmoid(r) * wkv and the bf16 downconvert for the Wo GEMM.
// ---------------------------------------------------------------------------
__global__ __launch_bounds__(256) void wkv_scan_kernel(
    const float* __restrict__ k, const float* __restrict__ v,
    const float* __restrict__ r,
    const float* __restrict__ w_raw, const float* __restrict__ u,
    unsigned short* __restrict__ rwkv_bf)
{
    int gid = blockIdx.x * 256 + threadIdx.x;   // 0 .. B*C-1
    int b  = gid >> 10;                         // / Ck
    int ch = gid & (Ck - 1);
    float w  = -expf(w_raw[ch]);
    float uu = u[ch];
    float a = 0.f, bb = 0.f, p = -1e38f;
    size_t base = (size_t)b * Tk * Ck + ch;
    for (int t = 0; t < Tk; ++t) {
        size_t idx = base + (size_t)t * Ck;
        float kt = k[idx], vt = v[idx];
        float q  = fmaxf(p, uu + kt);
        float e1 = expf(p - q);
        float e2 = expf(uu + kt - q);
        float y  = (e1 * a + e2 * vt) / (e1 * bb + e2);
        float q2 = fmaxf(p + w, kt);
        float f1 = expf(p + w - q2);
        float f2 = expf(kt - q2);
        a = f1 * a + f2 * vt;
        bb = f1 * bb + f2;
        p = q2;
        float sr = 1.0f / (1.0f + expf(-r[idx]));
        rwkv_bf[idx] = f2bf(sr * y);
    }
}

// ---------------------------------------------------------------------------
// bf16 WMMA GEMM with double-buffered async global->LDS staging.
// out[M,N] = A[M,K] * W[N,K]^T   (W row-major [N,K] == WMMA B column-major)
// Block tile 128(M) x 256(N), K-step 32; 8 waves, 64x64 per wave (16 wmma/step).
// MODE 0: outf = acc
// MODE 1: outf = res + acc                      (Wo residual)
// MODE 2: outbf = bf16(relu(acc)^2)             (Fk -> kk)
// MODE 3: outf = res + sigmoid(rr) * acc        (Fv -> final output)
// ---------------------------------------------------------------------------
static constexpr int LD_T = 40;                 // LDS row stride (elems), pad 32->40

template<int MODE>
__global__ __launch_bounds__(256) void gemm_bf16_wmma(
    const unsigned short* __restrict__ A,
    const unsigned short* __restrict__ W,
    int N, int K,
    float* __restrict__ outf,
    unsigned short* __restrict__ outbf,
    const float* __restrict__ res,
    const float* __restrict__ rr)
{
    __shared__ unsigned short sA[2][128 * LD_T];    // 2 x 10,240 B
    __shared__ unsigned short sB[2][256 * LD_T];    // 2 x 20,480 B

    const int tid  = threadIdx.x;
    const int lane = tid & 31;
    const int wave = tid >> 5;
    const int wm = wave & 1;
    const int wn = wave >> 1;
    const int bm = blockIdx.y * 128;
    const int bn = blockIdx.x * 256;

    const int lrow = lane & 15;                 // row within 16 (A: M, B: N)
    const int akg  = (lane >> 4) << 3;          // A lane-group K offset: 0 / 8
    const int bkg  = (lane >> 4) << 3;          // B lane-group K offset: 0 / 8

    auto load_tile = [&](int buf, int k0) {
        #pragma unroll
        for (int c = 0; c < 2; ++c) {
            int q   = tid * 2 + c;
            int row = q >> 2;                   // 0..127
            int cch = (q & 3) << 3;             // 0,8,16,24
            async_cp16(&sA[buf][row * LD_T + cch],
                       A + (size_t)(bm + row) * K + (k0 + cch));
        }
        #pragma unroll
        for (int c = 0; c < 4; ++c) {
            int cch = c << 3;                   // 0,8,16,24
            async_cp16(&sB[buf][tid * LD_T + cch],
                       W + (size_t)(bn + tid) * K + (k0 + cch));
        }
    };

    v8f acc[4][4] = {};

    const int nk = K >> 5;
    load_tile(0, 0);

    for (int kt = 0; kt < nk; ++kt) {
        const int cur = kt & 1;
        if (kt + 1 < nk) {
            load_tile(cur ^ 1, (kt + 1) << 5);
            wait_asynccnt<6>();                 // previous 6 done, next 6 in flight
        } else {
            wait_asynccnt<0>();
        }
        __syncthreads();                        // tile visible to all waves

        const unsigned short* aB = &sA[cur][0];
        const unsigned short* bB = &sB[cur][0];
        bfrag16 af[4], bf[4];
        #pragma unroll
        for (int i = 0; i < 4; ++i) {
            int r = (wm * 64 + i * 16 + lrow) * LD_T;
            bhalf8 lo = *(const bhalf8*)(aB + r + akg);         // K: akg+0..7
            bhalf8 hi = *(const bhalf8*)(aB + r + akg + 16);    // K: akg+16..23
            af[i] = __builtin_shufflevector(lo, hi, 0,1,2,3,4,5,6,7,8,9,10,11,12,13,14,15);
        }
        #pragma unroll
        for (int j = 0; j < 4; ++j) {
            int r = (wn * 64 + j * 16 + lrow) * LD_T;
            bhalf8 lo = *(const bhalf8*)(bB + r + bkg);         // K: bkg+0..7
            bhalf8 hi = *(const bhalf8*)(bB + r + bkg + 16);    // K: bkg+16..23
            bf[j] = __builtin_shufflevector(lo, hi, 0,1,2,3,4,5,6,7,8,9,10,11,12,13,14,15);
        }
        #pragma unroll
        for (int i = 0; i < 4; ++i)
            #pragma unroll
            for (int j = 0; j < 4; ++j)
                acc[i][j] = __builtin_amdgcn_wmma_f32_16x16x32_bf16(
                    false, af[i], false, bf[j], (short)0, acc[i][j], false, false);

        __syncthreads();                        // all reads done before overwrite
    }

    const int cn  = lane & 15;
    const int cmh = (lane >> 4) << 3;
    #pragma unroll
    for (int i = 0; i < 4; ++i) {
        #pragma unroll
        for (int j = 0; j < 4; ++j) {
            #pragma unroll
            for (int e = 0; e < 8; ++e) {
                int m = bm + wm * 64 + i * 16 + cmh + e;
                int n = bn + wn * 64 + j * 16 + cn;
                size_t idx = (size_t)m * N + n;
                float vv = acc[i][j][e];
                if constexpr (MODE == 0) {
                    outf[idx] = vv;
                } else if constexpr (MODE == 1) {
                    outf[idx] = res[idx] + vv;
                } else if constexpr (MODE == 2) {
                    float tpos = fmaxf(vv, 0.0f);
                    outbf[idx] = f2bf(tpos * tpos);
                } else {
                    float sg = 1.0f / (1.0f + expf(-rr[idx]));
                    outf[idx] = res[idx] + sg * vv;
                }
            }
        }
    }
}

// ---------------------------------------------------------------------------
// host-side launch
// ---------------------------------------------------------------------------
extern "C" void kernel_launch(void* const* d_in, const int* in_sizes, int n_in,
                              void* d_out, int out_size, void* d_ws, size_t ws_size,
                              hipStream_t stream) {
    (void)in_sizes; (void)n_in; (void)out_size; (void)ws_size;

    const float* x          = (const float*)d_in[0];
    const float* ln1_g      = (const float*)d_in[1];
    const float* ln1_b      = (const float*)d_in[2];
    const float* ln2_g      = (const float*)d_in[3];
    const float* ln2_b      = (const float*)d_in[4];
    const float* time_decay = (const float*)d_in[5];
    const float* time_first = (const float*)d_in[6];
    const float* tmk        = (const float*)d_in[7];
    const float* tmv        = (const float*)d_in[8];
    const float* tmr        = (const float*)d_in[9];
    const float* Wk         = (const float*)d_in[10];
    const float* Wv         = (const float*)d_in[11];
    const float* Wr         = (const float*)d_in[12];
    const float* Wo         = (const float*)d_in[13];
    const float* fmk        = (const float*)d_in[14];
    const float* fmr        = (const float*)d_in[15];
    const float* Fk         = (const float*)d_in[16];
    const float* Fr         = (const float*)d_in[17];
    const float* Fv         = (const float*)d_in[18];
    float* out = (float*)d_out;

    const size_t CC  = (size_t)Ck * Ck;        // 1M
    const size_t CH  = (size_t)Ck * HID;       // 4M
    const size_t BTC = (size_t)BT * Ck;        // 16M

    // workspace layout (bf16 = unsigned short, fp32 = float)
    unsigned short* wk_bf = (unsigned short*)d_ws;
    unsigned short* wv_bf = wk_bf + CC;
    unsigned short* wr_bf = wv_bf + CC;
    unsigned short* wo_bf = wr_bf + CC;
    unsigned short* fr_w  = wo_bf + CC;
    unsigned short* fk_w  = fr_w  + CC;
    unsigned short* fv_w  = fk_w  + CH;
    unsigned short* xk_bf = fv_w  + CH;        // reused later as rwkv_bf
    unsigned short* xv_bf = xk_bf + BTC;       // reused later as fk_bf
    unsigned short* xr_bf = xv_bf + BTC;       // reused later as fr_bf
    float* k_f  = (float*)(xr_bf + BTC);       // reused later as rr_f
    float* v_f  = k_f  + BTC;
    float* r_f  = v_f  + BTC;
    float* x1_f = r_f  + BTC;
    unsigned short* kk_bf = (unsigned short*)(x1_f + BTC);

    unsigned short* rwkv_bf = xk_bf;
    unsigned short* fk_bf   = xv_bf;
    unsigned short* fr_bf   = xr_bf;
    float*          rr_f    = k_f;

    // 1) weights fp32 -> bf16 (all weights are [N,K] row-major == WMMA B col-major)
    auto cvt = [&](const float* s, unsigned short* d, size_t n) {
        cvt_f32_bf16<<<(unsigned)((n + 255) / 256), 256, 0, stream>>>(s, d, (int)n);
    };
    cvt(Wk, wk_bf, CC); cvt(Wv, wv_bf, CC); cvt(Wr, wr_bf, CC); cvt(Wo, wo_bf, CC);
    cvt(Fr, fr_w,  CC); cvt(Fk, fk_w,  CH); cvt(Fv, fv_w,  CH);

    // 2) ln1 + time-shift + mixes -> xk,xv,xr (bf16)
    ln_mix_kernel<3><<<BT, 256, 0, stream>>>(x, ln1_g, ln1_b, tmk, tmv, tmr,
                                             xk_bf, xv_bf, xr_bf);

    // 3) k / v / r GEMMs (M=BT, N=C, K=C)
    dim3 gC(Ck / 256, BT / 128);
    gemm_bf16_wmma<0><<<gC, 256, 0, stream>>>(xk_bf, wk_bf, Ck, Ck, k_f, nullptr, nullptr, nullptr);
    gemm_bf16_wmma<0><<<gC, 256, 0, stream>>>(xv_bf, wv_bf, Ck, Ck, v_f, nullptr, nullptr, nullptr);
    gemm_bf16_wmma<0><<<gC, 256, 0, stream>>>(xr_bf, wr_bf, Ck, Ck, r_f, nullptr, nullptr, nullptr);

    // 4) WKV scan (+ sigmoid(r)*wkv fused, bf16 out)
    wkv_scan_kernel<<<(Bk * Ck) / 256, 256, 0, stream>>>(k_f, v_f, r_f,
                                                         time_decay, time_first, rwkv_bf);

    // 5) x1 = x + rwkv @ Wo^T
    gemm_bf16_wmma<1><<<gC, 256, 0, stream>>>(rwkv_bf, wo_bf, Ck, Ck, x1_f, nullptr, x, nullptr);

    // 6) ln2 + time-shift + mixes -> fk, fr (bf16)
    ln_mix_kernel<2><<<BT, 256, 0, stream>>>(x1_f, ln2_g, ln2_b, fmk, fmr, nullptr,
                                             fk_bf, fr_bf, nullptr);

    // 7) kk = bf16(relu(fk @ Fk^T)^2)   (M=BT, N=HID, K=C)
    dim3 gH(HID / 256, BT / 128);
    gemm_bf16_wmma<2><<<gH, 256, 0, stream>>>(fk_bf, fk_w, HID, Ck, nullptr, kk_bf, nullptr, nullptr);

    // 8) rr = fr @ Fr^T
    gemm_bf16_wmma<0><<<gC, 256, 0, stream>>>(fr_bf, fr_w, Ck, Ck, rr_f, nullptr, nullptr, nullptr);

    // 9) out = x1 + sigmoid(rr) * (kk @ Fv^T)   (M=BT, N=C, K=HID)
    gemm_bf16_wmma<3><<<gC, 256, 0, stream>>>(kk_bf, fv_w, Ck, HID, out, nullptr, x1_f, rr_f);
}
